// DendriteLayer_19602230739126
// MI455X (gfx1250) — matrix operand order repaired and checked by reading, not compiled, not run on hardware
//
#include <hip/hip_runtime.h>
#include <hip/hip_bf16.h>

typedef __bf16 bf16_t;
typedef __attribute__((ext_vector_type(16))) __bf16 v16bf;
typedef __attribute__((ext_vector_type(4)))  __bf16 v4bf;
typedef __attribute__((ext_vector_type(8)))  float  v8f;

#define IN_DIM  1024
#define N_DEND  8192
#define BATCH   4096
#define OUT_DIM 512
#define DPC     16
#define KWIN    819

// GEMM tiling: block tile 128x128, 8 waves (wave32), wave tile 32x64,
// K-step 32 (one v_wmma_f32_16x16x32_bf16 per 16x16x32 sub-problem).
#define BM 128
#define BN 128
#define BK 32
#define LDSS 40   // bf16 elems per LDS row (32 + 8 pad to break bank conflicts)

#if __has_builtin(__builtin_amdgcn_global_load_async_to_lds_b128) && \
    __has_builtin(__builtin_amdgcn_s_wait_asynccnt)
#  define USE_ASYNC_LDS 1
// The builtin's pointer params are int4-typed, global (AS1) / LDS (AS3).
typedef __attribute__((__vector_size__(4 * sizeof(int)))) int v4i;
typedef __attribute__((address_space(1))) v4i* gv4i_p;
typedef __attribute__((address_space(3))) v4i* lv4i_p;
#else
#  define USE_ASYNC_LDS 0
#endif

// ---------------------------------------------------------------------------
// 1) x (fp32) -> bf16
__global__ void cvt_x_kernel(const float* __restrict__ x, bf16_t* __restrict__ xb, int n) {
    int i = (blockIdx.x * blockDim.x + threadIdx.x) * 4;
    if (i + 3 >= n) return;
    float4 v = *(const float4*)(x + i);
    v4bf o;
    o[0] = (bf16_t)v.x; o[1] = (bf16_t)v.y; o[2] = (bf16_t)v.z; o[3] = (bf16_t)v.w;
    *(v4bf*)(xb + i) = o;
}

// 2) (w1 * mask) -> bf16, fused once per launch
__global__ void cvt_w1_kernel(const float* __restrict__ w1, const unsigned char* __restrict__ m,
                              bf16_t* __restrict__ w1b, int n) {
    int i = (blockIdx.x * blockDim.x + threadIdx.x) * 4;
    if (i + 3 >= n) return;
    float4 v = *(const float4*)(w1 + i);
    uchar4 mm = *(const uchar4*)(m + i);
    v4bf o;
    o[0] = (bf16_t)(mm.x ? v.x : 0.f);
    o[1] = (bf16_t)(mm.y ? v.y : 0.f);
    o[2] = (bf16_t)(mm.z ? v.z : 0.f);
    o[3] = (bf16_t)(mm.w ? v.w : 0.f);
    *(v4bf*)(w1b + i) = o;
}

// ---------------------------------------------------------------------------
// 3) H = x_bf16 @ w1_bf16^T + b1, WMMA bf16 -> f32, double-buffered LDS with
//    CDNA5 async global->LDS copies (ASYNCcnt) when the toolchain exposes them.
__global__ __launch_bounds__(256)
void gemm_h_kernel(const bf16_t* __restrict__ A, const bf16_t* __restrict__ B,
                   const float* __restrict__ b1, float* __restrict__ H) {
    __shared__ __align__(16) bf16_t shA[2][BM * LDSS];
    __shared__ __align__(16) bf16_t shB[2][BN * LDSS];

    const int tid   = threadIdx.x;
    const int lane  = tid & 31;
    const int wave  = tid >> 5;
    const int l     = lane & 15;       // column / row-within-halfwave
    const int hi    = lane >> 4;       // half-wave select
    const int waveM = wave >> 1;       // 0..3
    const int waveN = wave & 1;        // 0..1
    const int blockM = blockIdx.y * BM;
    const int blockN = blockIdx.x * BN;

    // Each thread owns two fixed 16B chunks per matrix per K-step.
    const int c0 = tid, c1 = tid + 256;
    const int r0 = c0 >> 2, kp0 = (c0 & 3) * 8;
    const int r1 = c1 >> 2, kp1 = (c1 & 3) * 8;
    const bf16_t* gA0 = A + (size_t)(blockM + r0) * IN_DIM + kp0;
    const bf16_t* gA1 = A + (size_t)(blockM + r1) * IN_DIM + kp1;
    const bf16_t* gB0 = B + (size_t)(blockN + r0) * IN_DIM + kp0;
    const bf16_t* gB1 = B + (size_t)(blockN + r1) * IN_DIM + kp1;
    const int dA0 = r0 * LDSS + kp0, dA1 = r1 * LDSS + kp1;

    v8f acc[2][4];
    const v8f vzero = {0.f,0.f,0.f,0.f,0.f,0.f,0.f,0.f};
    for (int mi = 0; mi < 2; ++mi)
        for (int ni = 0; ni < 4; ++ni)
            acc[mi][ni] = vzero;

    auto issue_tile = [&](int kk, int buf) {
#if USE_ASYNC_LDS
        // 4 x global_load_async_to_lds_b128 per thread, tracked by ASYNCcnt
        __builtin_amdgcn_global_load_async_to_lds_b128(
            (gv4i_p)(gA0 + kk), (lv4i_p)&shA[buf][dA0], 0, 0);
        __builtin_amdgcn_global_load_async_to_lds_b128(
            (gv4i_p)(gA1 + kk), (lv4i_p)&shA[buf][dA1], 0, 0);
        __builtin_amdgcn_global_load_async_to_lds_b128(
            (gv4i_p)(gB0 + kk), (lv4i_p)&shB[buf][dA0], 0, 0);
        __builtin_amdgcn_global_load_async_to_lds_b128(
            (gv4i_p)(gB1 + kk), (lv4i_p)&shB[buf][dA1], 0, 0);
#else
        *(uint4*)&shA[buf][dA0] = *(const uint4*)(gA0 + kk);
        *(uint4*)&shA[buf][dA1] = *(const uint4*)(gA1 + kk);
        *(uint4*)&shB[buf][dA0] = *(const uint4*)(gB0 + kk);
        *(uint4*)&shB[buf][dA1] = *(const uint4*)(gB1 + kk);
#endif
    };

    const int NT = IN_DIM / BK;        // 32 K-steps
    issue_tile(0, 0);

    for (int t = 0; t < NT; ++t) {
        const int buf = t & 1;
        // Software pipeline: kick off tile t+1 into the other buffer (that
        // buffer was last read at tile t-1; the trailing barrier of t-1
        // guarantees all waves are done with it).
        if (t + 1 < NT) issue_tile((t + 1) * BK, buf ^ 1);
#if USE_ASYNC_LDS
        // Async loads complete in order: allowing 4 outstanding means the
        // 4 we just issued for t+1 may float, but tile t's data is in LDS.
        if (t + 1 < NT) __builtin_amdgcn_s_wait_asynccnt(4);
        else            __builtin_amdgcn_s_wait_asynccnt(0);
#endif
        __syncthreads();

        // Fragment loads per the CDNA5 VGPR layout tables (CDNA5 ISA §7.12.2)
        union Frag { uint4 u[2]; v16bf v; };
        Frag a[2], b[4];
        for (int mi = 0; mi < 2; ++mi) {
            // A 16x32 bf16: lanes 0-15 row M=l hold K0-7 (V0-3) + K16-23 (V4-7);
            // lanes 16-31 hold K8-15 + K24-31.
            const bf16_t* p = &shA[buf][(waveM * 32 + mi * 16 + l) * LDSS];
            a[mi].u[0] = *(const uint4*)&p[hi * 8];
            a[mi].u[1] = *(const uint4*)&p[16 + hi * 8];
        }
        for (int ni = 0; ni < 4; ++ni) {
            // B 32x16 bf16: lanes 0-15 col N=l hold K0-7 + K16-23; lanes 16-31 K8-15 + K24-31.
            const bf16_t* p = &shB[buf][(waveN * 64 + ni * 16 + l) * LDSS];
            b[ni].u[0] = *(const uint4*)&p[hi * 8];
            b[ni].u[1] = *(const uint4*)&p[16 + hi * 8];
        }
        for (int mi = 0; mi < 2; ++mi)
            for (int ni = 0; ni < 4; ++ni)
                acc[mi][ni] = __builtin_amdgcn_wmma_f32_16x16x32_bf16(
                    false, a[mi].v, false, b[ni].v,
                    (short)0, acc[mi][ni], false, false);
        __syncthreads();
    }

    // Epilogue: C/D layout -> lane<16: (M=r, N=l); lane>=16: (M=8+r, N=l)
    for (int mi = 0; mi < 2; ++mi) {
        for (int ni = 0; ni < 4; ++ni) {
            int col = blockN + waveN * 64 + ni * 16 + l;
            float bias = b1[col];
            #pragma unroll
            for (int r = 0; r < 8; ++r) {
                int row = blockM + waveM * 32 + mi * 16 + hi * 8 + r;
                H[(size_t)row * N_DEND + col] = acc[mi][ni][r] + bias;
            }
        }
    }
}

// ---------------------------------------------------------------------------
// 4) Per-row k-th largest value via 4-pass 8-bit radix select on
//    order-preserving uint keys. One workgroup per row.
__global__ void topk_thresh_kernel(const float* __restrict__ H, float* __restrict__ thresh) {
    const int row = blockIdx.x;
    const float* hrow = H + (size_t)row * N_DEND;
    __shared__ unsigned int hist[256];
    __shared__ unsigned int s_prefix;
    __shared__ unsigned int s_remaining;
    if (threadIdx.x == 0) { s_prefix = 0u; s_remaining = KWIN; }
    __syncthreads();

    for (int pass = 3; pass >= 0; --pass) {
        hist[threadIdx.x] = 0u;
        __syncthreads();
        unsigned int prefix = s_prefix;
        unsigned int pmask  = (pass == 3) ? 0u : (0xFFFFFFFFu << ((pass + 1) * 8));
        for (int i = threadIdx.x; i < N_DEND; i += 256) {
            unsigned int u = __float_as_uint(hrow[i]);
            unsigned int key = (u & 0x80000000u) ? ~u : (u | 0x80000000u);
            if ((key & pmask) == prefix)
                atomicAdd(&hist[(key >> (pass * 8)) & 0xFF], 1u);
        }
        __syncthreads();
        if (threadIdx.x == 0) {
            unsigned int rem = s_remaining;
            unsigned int chosen = 0;
            for (int bin = 255; bin >= 0; --bin) {
                unsigned int c = hist[bin];
                if (c >= rem) { chosen = (unsigned int)bin; break; }
                rem -= c;
            }
            s_remaining = rem;
            s_prefix = prefix | (chosen << (pass * 8));
        }
        __syncthreads();
    }
    if (threadIdx.x == 0) {
        unsigned int key = s_prefix;
        thresh[row] = (key & 0x80000000u) ? __uint_as_float(key & 0x7FFFFFFFu)
                                          : __uint_as_float(~key);
    }
}

// ---------------------------------------------------------------------------
// 5) out[b,o] = sum_d thresh_mask(h[b, o*16+d]) * w2[o, o*16+d] + b2[o]
__global__ void out_kernel(const float* __restrict__ H, const float* __restrict__ thresh,
                           const float* __restrict__ w2, const float* __restrict__ b2,
                           float* __restrict__ out) {
    int gid = blockIdx.x * blockDim.x + threadIdx.x;
    if (gid >= BATCH * OUT_DIM) return;
    int b = gid >> 9;      // / OUT_DIM
    int o = gid & 511;     // % OUT_DIM
    float t = thresh[b];
    const float* hp = H  + (size_t)b * N_DEND + (size_t)o * DPC;
    const float* wp = w2 + (size_t)o * (OUT_DIM * DPC) + (size_t)o * DPC;
    float s = 0.f;
    #pragma unroll
    for (int d = 0; d < DPC; ++d) {
        float hv = hp[d];
        s += (hv >= t ? hv : 0.f) * wp[d];
    }
    out[gid] = s + b2[o];
}

// ---------------------------------------------------------------------------
extern "C" void kernel_launch(void* const* d_in, const int* in_sizes, int n_in,
                              void* d_out, int out_size, void* d_ws, size_t ws_size,
                              hipStream_t stream) {
    (void)in_sizes; (void)n_in; (void)out_size; (void)ws_size;
    const float*         x       = (const float*)d_in[0];
    const float*         w1      = (const float*)d_in[1];
    const float*         b1      = (const float*)d_in[2];
    const unsigned char* w1_mask = (const unsigned char*)d_in[3];
    const float*         w2      = (const float*)d_in[4];
    const float*         b2      = (const float*)d_in[5];
    float*               out     = (float*)d_out;

    char* ws = (char*)d_ws;
    bf16_t* xb  = (bf16_t*)ws;                                             //  8 MB
    bf16_t* w1b = (bf16_t*)(ws + (size_t)BATCH * IN_DIM * 2);              // 16 MB
    float*  H   = (float*)(ws + (size_t)BATCH * IN_DIM * 2
                              + (size_t)N_DEND * IN_DIM * 2);              // 128 MB
    float*  thr = (float*)((char*)H + (size_t)BATCH * N_DEND * 4);         // 16 KB

    cvt_x_kernel <<<(BATCH * IN_DIM / 4 + 255) / 256, 256, 0, stream>>>(x, xb, BATCH * IN_DIM);
    cvt_w1_kernel<<<(N_DEND * IN_DIM / 4 + 255) / 256, 256, 0, stream>>>(w1, w1_mask, w1b, N_DEND * IN_DIM);

    dim3 grid(N_DEND / BN, BATCH / BM);   // 64 x 32 blocks
    gemm_h_kernel<<<grid, 256, 0, stream>>>(xb, w1b, b1, H);

    topk_thresh_kernel<<<BATCH, 256, 0, stream>>>(H, thr);

    out_kernel<<<(BATCH * OUT_DIM + 255) / 256, 256, 0, stream>>>(H, thr, w2, b2, out);
}
